// PyConv2d_29326036697766
// MI455X (gfx1250) — matrix operand order repaired and checked by reading, not compiled, not run on hardware
//
#include <hip/hip_runtime.h>

// Implicit-GEMM 3x3 VALID conv for MI455X (gfx1250, wave32, WMMA).
// M = 256 (kout), N = 32*110*110 = 387200 (batch*spatial), K = 128*9 = 1152.
// Split-bf16 (hi/lo) with f32 accumulation via v_wmma_f32_16x16x32_bf16.
// Double-buffered LDS pipeline:
//   A-hi tile : TDM tensor_load_to_lds (TENSORcnt, wave 0 issues descriptor)
//   A-lo tile : global_load_async_to_lds_b128 (ASYNCcnt, per-lane)
//   B tile    : im2col gather + fp32->bf16 hi/lo split (VALU), LDS offset table

typedef __attribute__((ext_vector_type(16))) __bf16 v16bf;
typedef __attribute__((ext_vector_type(8)))  float  v8f;
typedef __attribute__((ext_vector_type(4)))  unsigned u32x4;
typedef __attribute__((ext_vector_type(8)))  int      i32x8;
typedef __attribute__((ext_vector_type(4)))  int      i32x4;

#define CIN   128
#define HIN   112
#define WIN   112
#define KOUT  256
#define PQ    110
#define KRED  1152            // CIN * 9
#define HW    (HIN * WIN)     // 12544
#define PQ2   (PQ * PQ)       // 12100
#define KCHUNKS 36            // 1152 / 32

__device__ __forceinline__ unsigned short bf16_hi_rne(float f) {
  unsigned u = __float_as_uint(f);
  u += 0x7FFFu + ((u >> 16) & 1u);          // round-to-nearest-even
  return (unsigned short)(u >> 16);
}
__device__ __forceinline__ float bf16_up(unsigned short h) {
  return __uint_as_float(((unsigned)h) << 16);
}

// ---------------------------------------------------------------------------
// TDM 2D tile load: global (bf16 matrix, row stride in elements) -> LDS.
// D# built per CDNA5 ISA ch.8: group0 = {flags, lds_addr, global_addr, type=2},
// group1 = {wg_mask/data_size, tensor dims, tile dims, dim0 stride}.
// clang-23 toolchain: 6-arg builtin (extra int32x8 descriptor words, zero here).
// ---------------------------------------------------------------------------
__device__ __forceinline__ void tdm_load_2d(unsigned lds_addr, const void* gptr,
                                            unsigned tile_w, unsigned tile_h,
                                            unsigned tensor_w, unsigned tensor_h,
                                            unsigned row_stride) {
  unsigned long long ga = (unsigned long long)gptr;
  u32x4 g0;
  g0[0] = 1u;                                            // count=1, user mode
  g0[1] = lds_addr;                                      // [63:32] LDS byte addr
  g0[2] = (unsigned)(ga & 0xFFFFFFFFu);                  // global_addr[31:0]
  g0[3] = (unsigned)((ga >> 32) & 0x01FFFFFFu) | (2u << 30);  // [56:32] | type=2
  i32x8 g1;
  g1[0] = (int)(1u << 16);                               // data_size=1 (2B), mask=0
  g1[1] = (int)((tensor_w & 0xFFFFu) << 16);             // tensor_dim0[15:0]
  g1[2] = (int)(((tensor_w >> 16) & 0xFFFFu) |
                ((tensor_h & 0xFFFFu) << 16));           // dim0[31:16] | dim1[15:0]
  g1[3] = (int)(((tensor_h >> 16) & 0xFFFFu) |
                ((tile_w & 0xFFFFu) << 16));             // dim1[31:16] | tile_dim0
  g1[4] = (int)(tile_h & 0xFFFFu);                       // tile_dim1 (tile_dim2=0)
  g1[5] = (int)row_stride;                               // tensor_dim0_stride[31:0]
  g1[6] = 0;                                             // stride[47:32], dim1_stride lo
  g1[7] = 0;
  i32x4 gz4 = {0, 0, 0, 0};
  i32x8 gz8 = {0, 0, 0, 0, 0, 0, 0, 0};
  __builtin_amdgcn_tensor_load_to_lds(g0, g1, gz4, gz4, gz8, 0);
}

// ---------------------------------------------------------------------------
// Pack weight OIHW fp32 -> bf16 hi/lo planes, layout [kout][c*9 + r*3 + s].
// ---------------------------------------------------------------------------
__global__ void pack_weight(const float* __restrict__ w,
                            unsigned short* __restrict__ whi,
                            unsigned short* __restrict__ wlo) {
  int idx = blockIdx.x * 256 + threadIdx.x;   // 0 .. 256*1152-1 exactly
  int ko  = idx / KRED;
  int kk  = idx - ko * KRED;
  int c   = kk / 9;
  int rs  = kk - c * 9;
  int r   = rs / 3;
  int s   = rs - r * 3;
  float v = w[(((size_t)ko * CIN + c) * 3 + r) * 3 + s];
  unsigned short hi = bf16_hi_rne(v);
  unsigned short lo = bf16_hi_rne(v - bf16_up(hi));
  whi[(size_t)ko * KRED + kk] = hi;
  wlo[(size_t)ko * KRED + kk] = lo;
}

// ---------------------------------------------------------------------------
// Main implicit-GEMM kernel. Block tile: 128 (kout) x 128 (spatial).
// 8 waves arranged 2 (M) x 4 (N); each wave: 64x32 = 8 accumulators 16x16.
// ---------------------------------------------------------------------------
__global__ __launch_bounds__(256)
void conv_igemm(const float* __restrict__ x,
                const unsigned short* __restrict__ whi,
                const unsigned short* __restrict__ wlo,
                float* __restrict__ out) {
  // [buf][row][32 k] bf16, row pitch 64B -> 32B-aligned fragment reads.
  __shared__ __attribute__((aligned(64))) unsigned short As_hi[2][128 * 32];
  __shared__ __attribute__((aligned(64))) unsigned short As_lo[2][128 * 32];
  __shared__ __attribute__((aligned(64))) unsigned short Bs_hi[2][128 * 32];
  __shared__ __attribute__((aligned(64))) unsigned short Bs_lo[2][128 * 32];
  __shared__ __attribute__((aligned(16))) int koffs[KRED];   // im2col k -> elem offset

  const int t    = threadIdx.x;
  const int lane = t & 31;
  const int wid  = t >> 5;
  const int wm   = wid & 1;    // 0..1 -> 64-row (kout) slice
  const int wn   = wid >> 1;   // 0..3 -> 32-col (spatial) slice

  const int nblock = blockIdx.x;   // 0..3024  (387200/128)
  const int mblock = blockIdx.y;   // 0..1     (256/128)

  // Build k -> (c*HW + r*WIN + s) offset table once per block.
  for (int kg = t; kg < KRED; kg += 256) {
    int c  = kg / 9;
    int rs = kg - c * 9;
    int r  = rs / 3;
    int s  = rs - r * 3;
    koffs[kg] = c * HW + r * WIN + s;
  }

  // Staging assignment: thread covers 16 k for one row (128 rows x 2 halves).
  const int row   = t >> 1;          // 0..127
  const int khalf = (t & 1) * 16;    // 0 or 16

  // im2col coordinates for the B (input) row this thread stages.
  const int sg  = nblock * 128 + row;          // global spatial index
  const int b   = sg / PQ2;
  const int rm  = sg - b * PQ2;
  const int p   = rm / PQ;
  const int q   = rm - p * PQ;
  const float* xb = x + (size_t)b * CIN * HW + (size_t)p * WIN + q;

  const unsigned short* ghiT = whi + (size_t)(mblock * 128) * KRED;   // TDM tile base
  const unsigned short* glo  = wlo + (size_t)(mblock * 128 + row) * KRED + khalf;

  v8f acc[4][2] = {};

  // ---- A-hi: one TDM descriptor per chunk (wave 0); A-lo: per-lane async ----
  auto stageA = [&](int kc, int buf) {
    if (wid == 0) {
      tdm_load_2d((unsigned)(unsigned long long)&As_hi[buf][0],
                  ghiT + kc * 32, /*tile*/ 32, 128,
                  /*tensor*/ KRED, KOUT, /*stride*/ KRED);
    }
    unsigned long long gl = (unsigned long long)(glo + kc * 32);
    unsigned dl = (unsigned)(unsigned long long)&As_lo[buf][row * 32 + khalf];
    asm volatile("global_load_async_to_lds_b128 %0, %1, off"
                 :: "v"(dl), "v"(gl) : "memory");
    asm volatile("global_load_async_to_lds_b128 %0, %1, off offset:16"
                 :: "v"(dl), "v"(gl) : "memory");
  };

  // ---- B staging: table-driven gather + fp32 -> bf16 hi/lo split ----
  auto stageB = [&](int kc, int buf) {
    const int base = kc * 32 + khalf;
    int offs[16];
#pragma unroll
    for (int j = 0; j < 4; ++j)
      ((uint4*)offs)[j] = ((const uint4*)&koffs[base])[j];
#pragma unroll
    for (int i = 0; i < 16; ++i) {
      const float v = xb[offs[i]];                 // always in-bounds (VALID conv)
      const unsigned short hi = bf16_hi_rne(v);
      Bs_hi[buf][row * 32 + khalf + i] = hi;
      Bs_lo[buf][row * 32 + khalf + i] = bf16_hi_rne(v - bf16_up(hi));
    }
  };

  auto waitMovers = []() {
    asm volatile("s_wait_asynccnt 0" ::: "memory");
    __builtin_amdgcn_s_wait_tensorcnt((short)0);   // non-issuing waves: cnt==0, no-op
  };

  __syncthreads();          // koffs table visible before first stageB

  // Prologue: fill buffer 0.
  stageA(0, 0);
  stageB(0, 0);
  waitMovers();
  __syncthreads();

  const int nrow0 = wn * 32 + (lane & 15);
  const int koffB = (lane < 16) ? 0 : 16;   // ushort offset into B row
  const int selA  = (lane < 16) ? 0 : 1;    // uint4 chunk select in A row

  for (int kc = 0; kc < KCHUNKS; ++kc) {
    const int cur = kc & 1;

    // Stage next chunk into the other buffer while we compute this one.
    if (kc + 1 < KCHUNKS) {
      stageA(kc + 1, cur ^ 1);
      stageB(kc + 1, cur ^ 1);
      if (kc + 2 < KCHUNKS) {
        __builtin_prefetch(ghiT + (size_t)row * KRED + (kc + 2) * 32, 0, 1);
        __builtin_prefetch(glo + (kc + 2) * 32, 0, 1);   // global_prefetch_b8
      }
    }

    // ---- B fragments: lane(0..15)=col n, K 0..15; lane(16..31)=K 16..31 ----
    v16bf bh[2], bl[2];
#pragma unroll
    for (int nt = 0; nt < 2; ++nt) {
      bh[nt] = *(const v16bf*)&Bs_hi[cur][(nrow0 + nt * 16) * 32 + koffB];
      bl[nt] = *(const v16bf*)&Bs_lo[cur][(nrow0 + nt * 16) * 32 + koffB];
    }

    // ---- A fragments + 3-term split-bf16 WMMA ----
#pragma unroll
    for (int mt = 0; mt < 4; ++mt) {
      const int arow = wm * 64 + mt * 16 + (lane & 15);
      union { uint4 u[2]; v16bf v; } ah, al;
      const uint4* rh = (const uint4*)&As_hi[cur][arow * 32];
      const uint4* rl = (const uint4*)&As_lo[cur][arow * 32];
      ah.u[0] = rh[2 * selA]; ah.u[1] = rh[2 * selA + 1];  // K 0..15 (or 16..31)
      al.u[0] = rl[2 * selA]; al.u[1] = rl[2 * selA + 1];
#pragma unroll
      for (int nt = 0; nt < 2; ++nt) {
        acc[mt][nt] = __builtin_amdgcn_wmma_f32_16x16x32_bf16(
            false, ah.v, false, bh[nt], (short)0, acc[mt][nt], false, false);
        acc[mt][nt] = __builtin_amdgcn_wmma_f32_16x16x32_bf16(
            false, ah.v, false, bl[nt], (short)0, acc[mt][nt], false, false);
        acc[mt][nt] = __builtin_amdgcn_wmma_f32_16x16x32_bf16(
            false, al.v, false, bh[nt], (short)0, acc[mt][nt], false, false);
      }
    }

    // Movers must land + everyone done with cur before the buffer flip.
    waitMovers();
    __syncthreads();
  }

  // ---- Epilogue: C/D layout -> NCHW fp32 stores ----
  // VGPR j: lanes 0-15 hold M=j, lanes 16-31 hold M=j+8; N = lane&15.
  const int mrow0 = mblock * 128 + wm * 64 + ((lane >> 4) << 3);
#pragma unroll
  for (int nt = 0; nt < 2; ++nt) {
    const int nn  = nblock * 128 + wn * 32 + nt * 16 + (lane & 15);
    const int ob  = nn / PQ2;
    const int orm = nn - ob * PQ2;
    const int op  = orm / PQ;
    const int oq  = orm - op * PQ;
    const size_t obase = (size_t)ob * KOUT * PQ2 + (size_t)op * PQ + oq;
#pragma unroll
    for (int mt = 0; mt < 4; ++mt) {
#pragma unroll
      for (int j = 0; j < 8; ++j) {
        out[obase + (size_t)(mrow0 + mt * 16 + j) * PQ2] = acc[mt][nt][j];
      }
    }
  }
}

// ---------------------------------------------------------------------------
extern "C" void kernel_launch(void* const* d_in, const int* in_sizes, int n_in,
                              void* d_out, int out_size, void* d_ws, size_t ws_size,
                              hipStream_t stream) {
  const float* x = (const float*)d_in[0];     // (32,128,112,112) fp32
  const float* w = (const float*)d_in[1];     // (256,128,3,3)    fp32
  float* out = (float*)d_out;                 // (32,256,110,110) fp32

  unsigned short* whi = (unsigned short*)d_ws;                 // 256*1152 bf16
  unsigned short* wlo = whi + (size_t)KOUT * KRED;             // 256*1152 bf16

  pack_weight<<<(KOUT * KRED) / 256, 256, 0, stream>>>(w, whi, wlo);

  dim3 grid(387200 / 128, KOUT / 128);   // (3025, 2)
  conv_igemm<<<grid, 256, 0, stream>>>(x, whi, wlo, out);
}
